// TABGNN2_82643760709804
// MI455X (gfx1250) — matrix operand order repaired and checked by reading, not compiled, not run on hardware
//
#include <hip/hip_runtime.h>
#include <hip/hip_bf16.h>
#include <cstddef>
#include <cstdint>

// ---------------- model dims ----------------
static constexpr int Hd      = 64;
static constexpr int Cd      = 8;
static constexpr int Bd      = 8;
static constexpr int Rd      = 512;
static constexpr int Nnodes  = 2048;
static constexpr int FFd     = 2048;
static constexpr int Eedges  = Bd * Rd;          // 4096
static constexpr int Mtab    = Bd * Rd * Cd;     // 32768 rows of H for tab encoders
static constexpr int TABELEMS = Mtab * Hd;       // 2097152

typedef __attribute__((ext_vector_type(16))) _Float16 v16h;
typedef __attribute__((ext_vector_type(2)))  _Float16 v2h;
typedef __attribute__((ext_vector_type(8)))  float    v8f;
typedef __attribute__((ext_vector_type(4)))  unsigned int v4u;
typedef __attribute__((ext_vector_type(8)))  int      v8i;
typedef __attribute__((ext_vector_type(4)))  int      v4i;

// =====================================================================
// WMMA GEMM:  Out[M,Nd] = act(A[M,K] * W[Nd,K]^T + bias[Nd])
// block = 128 threads (4 waves). Wave computes a 16(M) x 64(N) tile:
// one A fragment is reused across 4 B fragments -> 4 WMMAs per K-slice.
// B tile (W^T slice, 32x64) staged transposed as f16 in LDS, layout
// lb[k][n] so lane=n reads 16 halves at stride 64 (one per k) per 16-wide sub-tile
// (matches CDNA5 16-bit B operand layout).
// =====================================================================
template <int ACT>
__global__ __launch_bounds__(128) void k_gemm(const float* __restrict__ A,
                                              const float* __restrict__ W,
                                              const float* __restrict__ bias,
                                              float* __restrict__ Out,
                                              int M, int Nd, int K) {
  __shared__ _Float16 lb[32 * 64];
  const int tid  = threadIdx.x;
  const int lane = tid & 31;
  const int wave = tid >> 5;
  const int n0   = blockIdx.x * 64;
  const int m0   = (blockIdx.y * 4 + wave) * 16;
  const int m    = lane & 15;
  const int g    = lane >> 4;
  const float* __restrict__ arow = A + (size_t)(m0 + m) * K;

  v8f acc[4];
#pragma unroll
  for (int nt = 0; nt < 4; ++nt)
    acc[nt] = (v8f){0.f, 0.f, 0.f, 0.f, 0.f, 0.f, 0.f, 0.f};

  for (int k0 = 0; k0 < K; k0 += 32) {
    // ---- cooperative stage of W^T tile into LDS (f32 -> f16, transposed)
    // unit u covers 2 kk x 2 nn; packed v2h stores along n.
#pragma unroll
    for (int i = 0; i < 4; ++i) {
      int u  = tid + i * 128;        // 0..511
      int kk = (u & 15) * 2;         // 0..30 even
      int nn = (u >> 4) * 2;         // 0..62 even
      const float2 r0 = *(const float2*)(W + (size_t)(n0 + nn) * K + k0 + kk);
      const float2 r1 = *(const float2*)(W + (size_t)(n0 + nn + 1) * K + k0 + kk);
      v2h p0; p0[0] = (_Float16)r0.x; p0[1] = (_Float16)r1.x;
      v2h p1; p1[0] = (_Float16)r0.y; p1[1] = (_Float16)r1.y;
      *(v2h*)(&lb[(kk + 0) * 64 + nn]) = p0;
      *(v2h*)(&lb[(kk + 1) * 64 + nn]) = p1;
    }
    if (k0 + 32 < K) __builtin_prefetch(arow + k0 + 32, 0, 3);

    // ---- A fragment: lane(m,g): j0..7 -> K=k0+8g.. ; j8..15 -> K=k0+16+8g..
    float4 a0 = *(const float4*)(arow + k0 + 8 * g);
    float4 a1 = *(const float4*)(arow + k0 + 8 * g + 4);
    float4 a2 = *(const float4*)(arow + k0 + 16 + 8 * g);
    float4 a3 = *(const float4*)(arow + k0 + 16 + 8 * g + 4);
    v16h av;
    av[0] = (_Float16)a0.x;  av[1] = (_Float16)a0.y;
    av[2] = (_Float16)a0.z;  av[3] = (_Float16)a0.w;
    av[4] = (_Float16)a1.x;  av[5] = (_Float16)a1.y;
    av[6] = (_Float16)a1.z;  av[7] = (_Float16)a1.w;
    av[8] = (_Float16)a2.x;  av[9] = (_Float16)a2.y;
    av[10] = (_Float16)a2.z; av[11] = (_Float16)a2.w;
    av[12] = (_Float16)a3.x; av[13] = (_Float16)a3.y;
    av[14] = (_Float16)a3.z; av[15] = (_Float16)a3.w;

    __syncthreads();
#pragma unroll
    for (int nt = 0; nt < 4; ++nt) {
      v16h bv;
      const _Float16* bp = lb + nt * 16 + m;  // lane = N col of B tile; j -> K (table order)
#pragma unroll
      for (int j = 0; j < 16; ++j) bv[j] = bp[(8 * g + (j & 7) + ((j >> 3) << 4)) * 64];
      acc[nt] = __builtin_amdgcn_wmma_f32_16x16x32_f16(false, av, false, bv,
                                                       (short)0, acc[nt], false, false);
    }
    __syncthreads();
  }

  // ---- epilogue: C/D layout: lane -> n = lane&15, vgpr r -> row r+8g
#pragma unroll
  for (int nt = 0; nt < 4; ++nt) {
    const int ncol = n0 + nt * 16 + m;
    const float bb = bias ? bias[ncol] : 0.f;
#pragma unroll
    for (int r = 0; r < 8; ++r) {
      int row = m0 + r + 8 * g;
      float v = acc[nt][r] + bb;
      if (ACT) v = v > 0.f ? v : 0.f;
      Out[(size_t)row * Nd + ncol] = v;
    }
  }
}

static void launch_gemm(const float* A, const float* W, const float* bias,
                        float* Out, int M, int Nd, int K, int act,
                        hipStream_t s) {
  dim3 grid(Nd / 64, M / 64);
  if (act)
    k_gemm<1><<<grid, 128, 0, s>>>(A, W, bias, Out, M, Nd, K);
  else
    k_gemm<0><<<grid, 128, 0, s>>>(A, W, bias, Out, M, Nd, K);
}

// =====================================================================
// TDM-based GEMM for Nd=64, K=64: whole 64x64 f32 weight (16 KB) DMA'd
// into LDS via TENSOR_LOAD_TO_LDS (Tensor Data Mover), per-wave
// completion via s_wait_tensorcnt. Then 2 K-slices x 4 WMMAs per wave.
// =====================================================================
template <int ACT>
__global__ __launch_bounds__(128) void k_gemm_tdm(const float* __restrict__ A,
                                                  const float* __restrict__ W,
                                                  const float* __restrict__ bias,
                                                  float* __restrict__ Out, int M) {
  __shared__ __align__(16) float ldsW[64 * 64];   // row-major W[n][k]
  const int tid  = threadIdx.x;
  const int lane = tid & 31;
  const int wave = tid >> 5;
  const int m0   = (blockIdx.x * 4 + wave) * 16;
  const int m    = lane & 15;
  const int g    = lane >> 4;

  // ---- Tensor DMA descriptor (D#), 2D: tensor 64x64 f32, tile = whole
  unsigned long long ga = (unsigned long long)(uintptr_t)W;
  unsigned ldsoff = (unsigned)(uintptr_t)(&ldsW[0]);
  v4u g0;
  g0[0] = 1u;                                        // count=1, user mode
  g0[1] = ldsoff;                                    // lds_addr
  g0[2] = (unsigned)(ga & 0xffffffffu);              // global_addr[31:0]
  g0[3] = (unsigned)((ga >> 32) & 0x1ffffffu) | (2u << 30);  // addr[56:32] | type=2
  v8i g1;
  g1[0] = (int)(2u << 16);          // workgroup_mask=0, data_size=2 (4B)
  g1[1] = (int)(64u << 16);         // tensor_dim0[15:0]=64 in bits[31:16]
  g1[2] = (int)(64u << 16);         // dim0 hi=0 | tensor_dim1[15:0]=64
  g1[3] = (int)(64u << 16);         // dim1 hi=0 | tile_dim0=64
  g1[4] = (int)64u;                 // tile_dim1=64, tile_dim2=0
  g1[5] = (int)64u;                 // tensor_dim0_stride low = 64
  g1[6] = 0;                        // stride0 hi | stride1 lo
  g1[7] = 0;                        // stride1 hi
  v4i g2; g2[0] = 0; g2[1] = 0; g2[2] = 0; g2[3] = 0;
  v4i g3 = g2;
  v8i g4; g4[0] = 0; g4[1] = 0; g4[2] = 0; g4[3] = 0;
  g4[4] = 0; g4[5] = 0; g4[6] = 0; g4[7] = 0;
  __builtin_amdgcn_tensor_load_to_lds(g0, g1, g2, g3, g4, 0);
  __builtin_amdgcn_s_wait_tensorcnt(0);   // wave-local completion; LDS now valid

  const float* __restrict__ arow = A + (size_t)(m0 + m) * 64;
  v8f acc[4];
#pragma unroll
  for (int nt = 0; nt < 4; ++nt)
    acc[nt] = (v8f){0.f, 0.f, 0.f, 0.f, 0.f, 0.f, 0.f, 0.f};

#pragma unroll
  for (int ks = 0; ks < 2; ++ks) {
    const int k0 = ks * 32;
    float4 a0 = *(const float4*)(arow + k0 + 8 * g);
    float4 a1 = *(const float4*)(arow + k0 + 8 * g + 4);
    float4 a2 = *(const float4*)(arow + k0 + 16 + 8 * g);
    float4 a3 = *(const float4*)(arow + k0 + 16 + 8 * g + 4);
    v16h av;
    av[0] = (_Float16)a0.x;  av[1] = (_Float16)a0.y;
    av[2] = (_Float16)a0.z;  av[3] = (_Float16)a0.w;
    av[4] = (_Float16)a1.x;  av[5] = (_Float16)a1.y;
    av[6] = (_Float16)a1.z;  av[7] = (_Float16)a1.w;
    av[8] = (_Float16)a2.x;  av[9] = (_Float16)a2.y;
    av[10] = (_Float16)a2.z; av[11] = (_Float16)a2.w;
    av[12] = (_Float16)a3.x; av[13] = (_Float16)a3.y;
    av[14] = (_Float16)a3.z; av[15] = (_Float16)a3.w;
#pragma unroll
    for (int nt = 0; nt < 4; ++nt) {
      v16h bv;   // B[k][n] = W[n][k]: n = nt*16+m, j -> k (table order)
#pragma unroll
      for (int j = 0; j < 16; ++j)
        bv[j] = (_Float16)ldsW[(size_t)(nt * 16 + m) * 64 + k0 + 8 * g + (j & 7) + ((j >> 3) << 4)];
      acc[nt] = __builtin_amdgcn_wmma_f32_16x16x32_f16(false, av, false, bv,
                                                       (short)0, acc[nt], false, false);
    }
  }

#pragma unroll
  for (int nt = 0; nt < 4; ++nt) {
    const int ncol = nt * 16 + m;
    const float bb = bias ? bias[ncol] : 0.f;
#pragma unroll
    for (int r = 0; r < 8; ++r) {
      int row = m0 + r + 8 * g;
      float v = acc[nt][r] + bb;
      if (ACT) v = v > 0.f ? v : 0.f;
      Out[(size_t)row * 64 + ncol] = v;
    }
  }
}

static void launch_gemm64(const float* A, const float* W, const float* bias,
                          float* Out, int M, int act, hipStream_t s) {
  dim3 grid(M / 64);
  if (act)
    k_gemm_tdm<1><<<grid, 128, 0, s>>>(A, W, bias, Out, M);
  else
    k_gemm_tdm<0><<<grid, 128, 0, s>>>(A, W, bias, Out, M);
}

// =====================================================================
// elementwise / copy helpers
// =====================================================================
__global__ void k_copy(float* __restrict__ o, const float* __restrict__ i, int n) {
  int t = blockIdx.x * 256 + threadIdx.x;
  if (t < n) o[t] = i[t];
}
__global__ void k_avg(float* __restrict__ o, const float* __restrict__ a,
                      const float* __restrict__ b, int n) {
  int t = blockIdx.x * 256 + threadIdx.x;
  if (t < n) o[t] = (a[t] + b[t]) * 0.5f;
}
__global__ void k_fill(float* __restrict__ p, float v, int n) {
  int t = blockIdx.x * 256 + threadIdx.x;
  if (t < n) p[t] = v;
}
__global__ void k_axpyhalf(float* __restrict__ a, const float* __restrict__ u, int n) {
  int t = blockIdx.x * 256 + threadIdx.x;
  if (t < n) a[t] += 0.5f * u[t];
}
// [B,R,C,H] -> [B,C,R,H]
__global__ void k_transpose(float* __restrict__ dst, const float* __restrict__ src) {
  int t = blockIdx.x * 256 + threadIdx.x;
  if (t >= TABELEMS) return;
  int hh = t & 63;
  int q  = t >> 6;
  int r  = q & 511;
  int q2 = q >> 9;
  int c  = q2 & 7;
  int b  = q2 >> 3;
  dst[t] = src[(((size_t)(b * Rd + r) * Cd + c) * Hd) + hh];
}

// =====================================================================
// attention
// =====================================================================
__global__ void k_attn_row(const float* __restrict__ qkv, float* __restrict__ o,
                           int nSeq) {
  int t = blockIdx.x * 256 + threadIdx.x;
  if (t >= nSeq * 64) return;
  int seq = t >> 6;
  int hh  = (t >> 3) & 7;
  int qi  = t & 7;
  const float* base = qkv + (size_t)seq * 8 * 192;
  float q[8];
#pragma unroll
  for (int d = 0; d < 8; ++d) q[d] = base[qi * 192 + hh * 8 + d];
  float sc[8];
  float mx = -3.0e38f;
#pragma unroll
  for (int k = 0; k < 8; ++k) {
    float s = 0.f;
#pragma unroll
    for (int d = 0; d < 8; ++d) s += q[d] * base[k * 192 + 64 + hh * 8 + d];
    s *= 0.35355339059327f;
    sc[k] = s;
    mx = fmaxf(mx, s);
  }
  float sum = 0.f;
#pragma unroll
  for (int k = 0; k < 8; ++k) { sc[k] = __expf(sc[k] - mx); sum += sc[k]; }
  float inv = 1.f / sum;
#pragma unroll
  for (int d = 0; d < 8; ++d) {
    float a = 0.f;
#pragma unroll
    for (int k = 0; k < 8; ++k) a += sc[k] * base[k * 192 + 128 + hh * 8 + d];
    o[(size_t)(seq * 8 + qi) * 64 + hh * 8 + d] = a * inv;
  }
}

// one wave32 per (seq, head, query); seq length 512
__global__ __launch_bounds__(256) void k_attn_col(const float* __restrict__ qkv,
                                                  float* __restrict__ o, int nSeq) {
  int wid  = (blockIdx.x * 256 + threadIdx.x) >> 5;
  int lane = threadIdx.x & 31;
  int q    = wid & 511;
  int hh   = (wid >> 9) & 7;
  int seq  = wid >> 12;
  if (seq >= nSeq) return;
  const float* base = qkv + (size_t)seq * 512 * 192;
  float qv[8];
#pragma unroll
  for (int d = 0; d < 8; ++d) qv[d] = base[(size_t)q * 192 + hh * 8 + d];
  float sc[16];
  float mx = -3.0e38f;
#pragma unroll
  for (int kk = 0; kk < 16; ++kk) {
    const float* kp = base + (size_t)(kk * 32 + lane) * 192 + 64 + hh * 8;
    float s = 0.f;
#pragma unroll
    for (int d = 0; d < 8; ++d) s += qv[d] * kp[d];
    s *= 0.35355339059327f;
    sc[kk] = s;
    mx = fmaxf(mx, s);
  }
#pragma unroll
  for (int off = 16; off > 0; off >>= 1) mx = fmaxf(mx, __shfl_xor(mx, off, 32));
  float sum = 0.f;
  float oacc[8] = {0.f, 0.f, 0.f, 0.f, 0.f, 0.f, 0.f, 0.f};
#pragma unroll
  for (int kk = 0; kk < 16; ++kk) {
    float p = __expf(sc[kk] - mx);
    sum += p;
    const float* vp = base + (size_t)(kk * 32 + lane) * 192 + 128 + hh * 8;
#pragma unroll
    for (int d = 0; d < 8; ++d) oacc[d] += p * vp[d];
  }
#pragma unroll
  for (int off = 16; off > 0; off >>= 1) sum += __shfl_xor(sum, off, 32);
  float inv = 1.f / sum;
#pragma unroll
  for (int d = 0; d < 8; ++d) {
    float v = oacc[d];
#pragma unroll
    for (int off = 16; off > 0; off >>= 1) v += __shfl_xor(v, off, 32);
    if (lane == 0) o[((size_t)seq * 512 + q) * 64 + hh * 8 + d] = v * inv;
  }
}

// =====================================================================
// layernorm (wave per 64-wide row), optional residual; dst may alias res
// =====================================================================
__global__ __launch_bounds__(256) void k_ln(const float* __restrict__ a,
                                            const float* __restrict__ res,
                                            const float* __restrict__ g,
                                            const float* __restrict__ b,
                                            float* __restrict__ dst, int M) {
  int row  = (blockIdx.x * 256 + threadIdx.x) >> 5;
  int lane = threadIdx.x & 31;
  if (row >= M) return;
  size_t base = (size_t)row * 64;
  float v0 = a[base + lane]      + (res ? res[base + lane]      : 0.f);
  float v1 = a[base + lane + 32] + (res ? res[base + lane + 32] : 0.f);
  float s = v0 + v1;
#pragma unroll
  for (int off = 16; off > 0; off >>= 1) s += __shfl_xor(s, off, 32);
  float mu = s * (1.f / 64.f);
  float d0 = v0 - mu, d1 = v1 - mu;
  float vs = d0 * d0 + d1 * d1;
#pragma unroll
  for (int off = 16; off > 0; off >>= 1) vs += __shfl_xor(vs, off, 32);
  float rstd = rsqrtf(vs * (1.f / 64.f) + 1e-5f);
  dst[base + lane]      = d0 * rstd * g[lane]      + b[lane];
  dst[base + lane + 32] = d1 * rstd * g[lane + 32] + b[lane + 32];
}

// t[b,r,c,:] = (LN(row_out[b,r,c,:]) + LN(col_out[b,c,r,:]))/2   (wave per row)
__global__ __launch_bounds__(256) void k_combine(float* __restrict__ t,
                                                 const float* __restrict__ ro,
                                                 const float* __restrict__ co,
                                                 const float* __restrict__ rg,
                                                 const float* __restrict__ rb,
                                                 const float* __restrict__ cg,
                                                 const float* __restrict__ cb) {
  int i    = (blockIdx.x * 256 + threadIdx.x) >> 5;
  int lane = threadIdx.x & 31;
  if (i >= Mtab) return;
  int b = i >> 12, r = (i >> 3) & 511, c = i & 7;
  size_t rbase = (size_t)i * 64;
  size_t cbase = ((size_t)(b * Cd + c) * Rd + r) * 64;
#pragma unroll
  for (int which = 0; which < 2; ++which) {
    const float* src = which ? co : ro;
    size_t base = which ? cbase : rbase;
    float v0 = src[base + lane], v1 = src[base + lane + 32];
    float s = v0 + v1;
#pragma unroll
    for (int off = 16; off > 0; off >>= 1) s += __shfl_xor(s, off, 32);
    float mu = s * (1.f / 64.f);
    float d0 = v0 - mu, d1 = v1 - mu;
    float vs = d0 * d0 + d1 * d1;
#pragma unroll
    for (int off = 16; off > 0; off >>= 1) vs += __shfl_xor(vs, off, 32);
    float rstd = rsqrtf(vs * (1.f / 64.f) + 1e-5f);
    const float* g = which ? cg : rg;
    const float* bb = which ? cb : rb;
    float o0 = d0 * rstd * g[lane] + bb[lane];
    float o1 = d1 * rstd * g[lane + 32] + bb[lane + 32];
    if (which == 0) {
      t[rbase + lane] = o0;
      t[rbase + lane + 32] = o1;
    } else {
      t[rbase + lane]      = (t[rbase + lane] + o0) * 0.5f;
      t[rbase + lane + 32] = (t[rbase + lane + 32] + o1) * 0.5f;
    }
  }
}

// =====================================================================
// GNN pieces
// =====================================================================
__global__ void k_node_emb(const float* __restrict__ x, const float* __restrict__ w,
                           const float* __restrict__ b, float* __restrict__ h) {
  int t = blockIdx.x * 256 + threadIdx.x;
  if (t >= Nnodes * 64) return;
  int n = t >> 6, j = t & 63;
  h[t] = x[n] * w[j] + b[j];
}

__global__ void k_concat(float* __restrict__ out, const float* __restrict__ nodes,
                         const float* __restrict__ edata, const int* __restrict__ i1,
                         const int* __restrict__ i2, int E_) {
  int t = blockIdx.x * 256 + threadIdx.x;
  if (t >= E_ * 192) return;
  int e = t / 192, j = t - e * 192;
  float v;
  if (j < 64)       v = nodes[(size_t)i1[e] * 64 + j];
  else if (j < 128) v = nodes[(size_t)i2[e] * 64 + (j - 64)];
  else              v = edata[(size_t)e * 64 + (j - 128)];
  out[t] = v;
}

__device__ inline void atomicMaxF(float* addr, float val) {
  unsigned int* u = (unsigned int*)addr;
  unsigned int old = *u;
  while (__uint_as_float(old) < val) {
    unsigned int prev = atomicCAS(u, old, __float_as_uint(val));
    if (prev == old) break;
    old = prev;
  }
}
__device__ inline void atomicMinF(float* addr, float val) {
  unsigned int* u = (unsigned int*)addr;
  unsigned int old = *u;
  while (__uint_as_float(old) > val) {
    unsigned int prev = atomicCAS(u, old, __float_as_uint(val));
    if (prev == old) break;
    old = prev;
  }
}

__global__ void k_scatter(const float* __restrict__ m, const int* __restrict__ dsti,
                          float* __restrict__ sum, float* __restrict__ sum2,
                          float* __restrict__ mxb, float* __restrict__ mnb, int E_) {
  int t = blockIdx.x * 256 + threadIdx.x;
  if (t >= E_ * 64) return;
  int e = t >> 6, j = t & 63;
  int d = dsti[e];
  float v = m[t];
  atomicAdd(&sum[(size_t)d * 64 + j], v);
  atomicAdd(&sum2[(size_t)d * 64 + j], v * v);
  atomicMaxF(&mxb[(size_t)d * 64 + j], v);
  atomicMinF(&mnb[(size_t)d * 64 + j], v);
}

__global__ void k_deg(const int* __restrict__ dsti, float* __restrict__ cnt, int E_) {
  int t = blockIdx.x * 256 + threadIdx.x;
  if (t < E_) atomicAdd(&cnt[dsti[t]], 1.f);
}

__global__ __launch_bounds__(256) void k_delta(const float* __restrict__ cnt,
                                               float* __restrict__ acc, int Nn) {
  __shared__ float sh[256];
  int tid = threadIdx.x;
  float s = 0.f;
  for (int i = tid; i < Nn; i += 256) s += log1pf(cnt[i]);
  sh[tid] = s;
  __syncthreads();
  for (int o = 128; o > 0; o >>= 1) {
    if (tid < o) sh[tid] += sh[tid + o];
    __syncthreads();
  }
  if (tid == 0) acc[0] = sh[0] / (float)Nn;
}

__global__ void k_post(const float* __restrict__ h, const float* __restrict__ sum,
                       const float* __restrict__ sum2, const float* __restrict__ mxb,
                       const float* __restrict__ mnb, const float* __restrict__ cnt,
                       const float* __restrict__ acc, float* __restrict__ post) {
  int t = blockIdx.x * 256 + threadIdx.x;
  if (t >= Nnodes * 64) return;
  int n = t >> 6, j = t & 63;
  float c = cnt[n];
  float cc = fmaxf(c, 1.f);
  float mean = sum[t] / cc;
  float mean2 = sum2[t] / cc;
  float diff = mean2 - mean * mean;
  if (diff < 0.f) diff = 0.f;
  float sd = sqrtf(diff + 1e-5f);
  bool has = c > 0.f;
  float mx = has ? mxb[t] : 0.f;
  float mn = has ? mnb[t] : 0.f;
  float logd = logf(cc + 1.f);
  float delta = acc[0];
  float s1 = logd / delta, s2 = delta / logd;
  float* row = post + (size_t)n * 832;
  row[j] = h[t];
  row[64 + j] = mean;  row[128 + j] = mx;      row[192 + j] = mn;      row[256 + j] = sd;
  row[320 + j] = mean * s1; row[384 + j] = mx * s1; row[448 + j] = mn * s1; row[512 + j] = sd * s1;
  row[576 + j] = mean * s2; row[640 + j] = mx * s2; row[704 + j] = mn * s2; row[768 + j] = sd * s2;
}

__global__ __launch_bounds__(256) void k_bn_stats(const float* __restrict__ y,
                                                  float* __restrict__ mean,
                                                  float* __restrict__ var, int rows) {
  __shared__ float s1[256], s2[256];
  int f = blockIdx.x, tid = threadIdx.x;
  float a = 0.f, b = 0.f;
  for (int r = tid; r < rows; r += 256) {
    float v = y[(size_t)r * 64 + f];
    a += v;
    b += v * v;
  }
  s1[tid] = a;
  s2[tid] = b;
  __syncthreads();
  for (int o = 128; o > 0; o >>= 1) {
    if (tid < o) { s1[tid] += s1[tid + o]; s2[tid] += s2[tid + o]; }
    __syncthreads();
  }
  if (tid == 0) {
    float m = s1[0] / (float)rows;
    mean[f] = m;
    var[f] = s2[0] / (float)rows - m * m;
  }
}

__global__ void k_bn_apply(float* __restrict__ h, const float* __restrict__ y,
                           const float* __restrict__ mean, const float* __restrict__ var,
                           const float* __restrict__ g, const float* __restrict__ b,
                           int n) {
  int t = blockIdx.x * 256 + threadIdx.x;
  if (t >= n) return;
  int j = t & 63;
  float bn = (y[t] - mean[j]) * rsqrtf(var[j] + 1e-5f) * g[j] + b[j];
  if (bn < 0.f) bn = 0.f;
  h[t] = (h[t] + bn) * 0.5f;
}

__global__ void k_writeout(float* __restrict__ out, const float* __restrict__ h,
                           const float* __restrict__ ea, int nh, int ne) {
  int t = blockIdx.x * 256 + threadIdx.x;
  if (t < nh) out[t] = h[t];
  else if (t < nh + ne) out[t] = ea[t - nh];
}

// =====================================================================
// host orchestration
// =====================================================================
struct TFp { const float *in_w,*in_b,*out_w,*out_b,*w1,*b1,*w2,*b2,*ln1g,*ln1b,*ln2g,*ln2b; };
struct TABp { TFp row, col; const float *rng,*rnb,*cng,*cnb; };
struct GNNp { const float *enc_w,*enc_b,*pre_w,*pre_b,*post_w,*post_b,*lin_w,*lin_b,
                          *bn_g,*bn_b,*eu1_w,*eu1_b,*eu2_w,*eu2_b; };

struct WS {
  float *T, *A, *B, *QKV, *O, *FF, *EA, *HN, *ENC, *CAT, *POST, *MSG,
        *SUM, *SUM2, *MAXB, *MINB, *CNT, *ACC, *BNM, *BNV, *Y, *Y2;
};

static void run_encoder(float* X, int M, const TFp& p, bool col, const WS& w,
                        hipStream_t s) {
  launch_gemm(X, p.in_w, p.in_b, w.QKV, M, 192, 64, 0, s);
  if (col) {
    int waves = (M / 512) * 8 * 512;
    k_attn_col<<<waves / 8, 256, 0, s>>>(w.QKV, w.O, M / 512);
  } else {
    int th = (M / 8) * 64;
    k_attn_row<<<th / 256, 256, 0, s>>>(w.QKV, w.O, M / 8);
  }
  launch_gemm64(w.O, p.out_w, p.out_b, w.QKV, M, 0, s);  // TDM path; proj reuses QKV
  k_ln<<<M / 8, 256, 0, s>>>(w.QKV, X, p.ln1g, p.ln1b, X, M);
  const int CH = 4096;
  for (int c = 0; c < M / CH; ++c) {
    launch_gemm(X + (size_t)c * CH * 64, p.w1, p.b1, w.FF, CH, FFd, 64, 1, s);
    launch_gemm(w.FF, p.w2, p.b2, w.O + (size_t)c * CH * 64, CH, 64, FFd, 0, s);
  }
  k_ln<<<M / 8, 256, 0, s>>>(w.O, X, p.ln2g, p.ln2b, X, M);
}

static void run_gnn(float* h, float* ea, const GNNp& p, const int* srcI,
                    const int* dstI, const WS& w, hipStream_t s) {
  const int nf = Nnodes * 64;
  launch_gemm64(ea, p.enc_w, p.enc_b, w.ENC, Eedges, 0, s);   // TDM path
  k_concat<<<(Eedges * 192 + 255) / 256, 256, 0, s>>>(w.CAT, h, w.ENC, dstI, srcI, Eedges);
  launch_gemm(w.CAT, p.pre_w, p.pre_b, w.MSG, Eedges, 64, 192, 0, s);
  k_fill<<<nf / 256, 256, 0, s>>>(w.SUM, 0.f, nf);
  k_fill<<<nf / 256, 256, 0, s>>>(w.SUM2, 0.f, nf);
  k_fill<<<nf / 256, 256, 0, s>>>(w.MAXB, -3.0e38f, nf);
  k_fill<<<nf / 256, 256, 0, s>>>(w.MINB, 3.0e38f, nf);
  k_scatter<<<Eedges * 64 / 256, 256, 0, s>>>(w.MSG, dstI, w.SUM, w.SUM2, w.MAXB,
                                              w.MINB, Eedges);
  k_post<<<nf / 256, 256, 0, s>>>(h, w.SUM, w.SUM2, w.MAXB, w.MINB, w.CNT, w.ACC, w.POST);
  launch_gemm(w.POST, p.post_w, p.post_b, w.Y, Nnodes, 64, 832, 0, s);
  launch_gemm64(w.Y, p.lin_w, p.lin_b, w.Y2, Nnodes, 0, s);   // TDM path
  k_bn_stats<<<64, 256, 0, s>>>(w.Y2, w.BNM, w.BNV, Nnodes);
  k_bn_apply<<<nf / 256, 256, 0, s>>>(h, w.Y2, w.BNM, w.BNV, p.bn_g, p.bn_b, nf);
  k_concat<<<(Eedges * 192 + 255) / 256, 256, 0, s>>>(w.CAT, h, ea, srcI, dstI, Eedges);
  launch_gemm(w.CAT, p.eu1_w, p.eu1_b, w.MSG, Eedges, 64, 192, 1, s);
  launch_gemm64(w.MSG, p.eu2_w, p.eu2_b, w.ENC, Eedges, 0, s); // TDM path
  k_axpyhalf<<<Eedges * 64 / 256, 256, 0, s>>>(ea, w.ENC, Eedges * 64);
}

extern "C" void kernel_launch(void* const* d_in, const int* in_sizes, int n_in,
                              void* d_out, int out_size, void* d_ws, size_t ws_size,
                              hipStream_t stream) {
  (void)in_sizes; (void)out_size; (void)ws_size;
  const float* x = (const float*)d_in[0];
  const int* edge_index = (const int*)d_in[1];
  const float* edge_attr = (const float*)d_in[2];
  // d_in[3] = mask (all True) -- ignored

  // ---- parameter leaves, insertion order depth-first ----
  static const int NPAR = 4 + 2 * 28 + 2 * 14;  // 88
  static const int psz[NPAR] = {
      64, 64, 64 * 512, 64,
      // tab layer 0: row TF
      192 * 64, 192, 64 * 64, 64, 2048 * 64, 2048, 64 * 2048, 64, 64, 64, 64, 64,
      // tab layer 0: col TF
      192 * 64, 192, 64 * 64, 64, 2048 * 64, 2048, 64 * 2048, 64, 64, 64, 64, 64,
      64, 64, 64, 64,
      // tab layer 1: row TF
      192 * 64, 192, 64 * 64, 64, 2048 * 64, 2048, 64 * 2048, 64, 64, 64, 64, 64,
      // tab layer 1: col TF
      192 * 64, 192, 64 * 64, 64, 2048 * 64, 2048, 64 * 2048, 64, 64, 64, 64, 64,
      64, 64, 64, 64,
      // gnn layer 0
      64 * 64, 64, 64 * 192, 64, 64 * 832, 64, 64 * 64, 64, 64, 64, 64 * 192, 64, 64 * 64, 64,
      // gnn layer 1
      64 * 64, 64, 64 * 192, 64, 64 * 832, 64, 64 * 64, 64, 64, 64, 64 * 192, 64, 64 * 64, 64};
  const float* P[NPAR];
  if (n_in >= 4 + NPAR) {
    for (int i = 0; i < NPAR; ++i) P[i] = (const float*)d_in[4 + i];
  } else {
    const float* base = (const float*)d_in[4];
    size_t off = 0;
    for (int i = 0; i < NPAR; ++i) { P[i] = base + off; off += psz[i]; }
  }
  int cur = 0;
  const float* node_w = P[cur++]; const float* node_b = P[cur++];
  const float* edge_w = P[cur++]; const float* edge_b = P[cur++];
  TABp tab[2];
  for (int l = 0; l < 2; ++l) {
    TFp* tfs[2] = {&tab[l].row, &tab[l].col};
    for (int t = 0; t < 2; ++t) {
      TFp& f = *tfs[t];
      f.in_w = P[cur++]; f.in_b = P[cur++]; f.out_w = P[cur++]; f.out_b = P[cur++];
      f.w1 = P[cur++]; f.b1 = P[cur++]; f.w2 = P[cur++]; f.b2 = P[cur++];
      f.ln1g = P[cur++]; f.ln1b = P[cur++]; f.ln2g = P[cur++]; f.ln2b = P[cur++];
    }
    tab[l].rng = P[cur++]; tab[l].rnb = P[cur++];
    tab[l].cng = P[cur++]; tab[l].cnb = P[cur++];
  }
  GNNp gnn[2];
  for (int l = 0; l < 2; ++l) {
    GNNp& g = gnn[l];
    g.enc_w = P[cur++]; g.enc_b = P[cur++]; g.pre_w = P[cur++]; g.pre_b = P[cur++];
    g.post_w = P[cur++]; g.post_b = P[cur++]; g.lin_w = P[cur++]; g.lin_b = P[cur++];
    g.bn_g = P[cur++]; g.bn_b = P[cur++]; g.eu1_w = P[cur++]; g.eu1_b = P[cur++];
    g.eu2_w = P[cur++]; g.eu2_b = P[cur++];
  }

  // ---- workspace carve-up (floats) ----
  float* wsf = (float*)d_ws;
  WS w;
  size_t o = 0;
  w.T = wsf + o;    o += (size_t)TABELEMS;       // 2,097,152
  w.A = wsf + o;    o += (size_t)TABELEMS;
  w.B = wsf + o;    o += (size_t)TABELEMS;
  w.QKV = wsf + o;  o += (size_t)Mtab * 192;     // 6,291,456
  w.O = wsf + o;    o += (size_t)TABELEMS;
  w.FF = wsf + o;   o += (size_t)4096 * 2048;    // 8,388,608
  w.EA = wsf + o;   o += (size_t)Eedges * 64;
  w.HN = wsf + o;   o += (size_t)Nnodes * 64;
  w.ENC = wsf + o;  o += (size_t)Eedges * 64;
  w.CAT = wsf + o;  o += (size_t)Eedges * 192;
  w.POST = wsf + o; o += (size_t)Nnodes * 832;
  w.MSG = wsf + o;  o += (size_t)Eedges * 64;
  w.SUM = wsf + o;  o += (size_t)Nnodes * 64;
  w.SUM2 = wsf + o; o += (size_t)Nnodes * 64;
  w.MAXB = wsf + o; o += (size_t)Nnodes * 64;
  w.MINB = wsf + o; o += (size_t)Nnodes * 64;
  w.CNT = wsf + o;  o += (size_t)Nnodes;
  w.ACC = wsf + o;  o += 16;
  w.BNM = wsf + o;  o += 64;
  w.BNV = wsf + o;  o += 64;
  w.Y = wsf + o;    o += (size_t)Nnodes * 64;
  w.Y2 = wsf + o;   o += (size_t)Nnodes * 64;

  const int* srcI = edge_index;
  const int* dstI = edge_index + Eedges;

  // ---- tabular transformer ----
  k_copy<<<(TABELEMS + 255) / 256, 256, 0, stream>>>(w.T, edge_attr, TABELEMS);
  for (int l = 0; l < 2; ++l) {
    k_copy<<<(TABELEMS + 255) / 256, 256, 0, stream>>>(w.A, w.T, TABELEMS);
    k_transpose<<<(TABELEMS + 255) / 256, 256, 0, stream>>>(w.B, w.T);
    run_encoder(w.A, Mtab, tab[l].row, false, w, stream);
    run_encoder(w.B, Mtab, tab[l].col, true, w, stream);
    k_combine<<<Mtab / 8, 256, 0, stream>>>(w.T, w.A, w.B, tab[l].rng, tab[l].rnb,
                                            tab[l].cng, tab[l].cnb);
  }

  // ---- edge embedding ----
  k_avg<<<(TABELEMS + 255) / 256, 256, 0, stream>>>(w.A, edge_attr, w.T, TABELEMS);
  launch_gemm(w.A, edge_w, edge_b, w.EA, Eedges, 64, 512, 0, stream);

  // ---- node embedding, degrees, delta ----
  k_node_emb<<<Nnodes * 64 / 256, 256, 0, stream>>>(x, node_w, node_b, w.HN);
  k_fill<<<(Nnodes + 255) / 256, 256, 0, stream>>>(w.CNT, 0.f, Nnodes);
  k_deg<<<(Eedges + 255) / 256, 256, 0, stream>>>(dstI, w.CNT, Eedges);
  k_delta<<<1, 256, 0, stream>>>(w.CNT, w.ACC, Nnodes);

  // ---- GNN layers ----
  for (int l = 0; l < 2; ++l) run_gnn(w.HN, w.EA, gnn[l], srcI, dstI, w, stream);

  // ---- outputs: (h [2048*64], ea [4096*64]) concatenated ----
  const int nh = Nnodes * 64, ne = Eedges * 64;
  k_writeout<<<(nh + ne + 255) / 256, 256, 0, stream>>>((float*)d_out, w.HN, w.EA, nh, ne);
}
